// ChempropBlock_9801115369512
// MI455X (gfx1250) — matrix operand order repaired and checked by reading, not compiled, not run on hardware
//
#include <hip/hip_runtime.h>
#include <hip/hip_bf16.h>

// Problem constants (fixed by the reference)
#define NV 10000
#define NE 160000
#define DIM 512
#define DEPTH 3

typedef __attribute__((ext_vector_type(16))) __bf16 v16bf;
typedef __attribute__((ext_vector_type(8)))  __bf16 v8bf;
typedef __attribute__((ext_vector_type(4)))  __bf16 v4bf;
typedef __attribute__((ext_vector_type(8)))  float  v8f;
typedef __attribute__((ext_vector_type(4)))  float  v4f;

// ---------------- elementwise / scatter kernels ----------------

__global__ __launch_bounds__(256) void k_zero(float* __restrict__ p, int n4) {
    int t = blockIdx.x * blockDim.x + threadIdx.x;
    if (t < n4) {
        v4f z = {};
        ((v4f*)p)[t] = z;
    }
}

// eh[e] = node_feats[src[e]] + edge_feats[e]   (eh lives in d_out)
// fused: node_msgs[dst[e]] += relu(eh[e])   (layer-0 segment sum; nm is L2-resident)
__global__ __launch_bounds__(256) void k_init_eh_scatter(const float* __restrict__ nf,
                                                         const float* __restrict__ ef,
                                                         const int* __restrict__ src,
                                                         const int* __restrict__ dst,
                                                         float* __restrict__ eh,
                                                         float* __restrict__ nm) {
    int t = blockIdx.x * blockDim.x + threadIdx.x;   // NE*128 threads (float4 units)
    int e = t >> 7, c = t & 127;
    v4f a = ((const v4f*)nf)[(size_t)src[e] * 128 + c];
    v4f b = ((const v4f*)ef)[(size_t)e * 128 + c];
    v4f s = a + b;
    ((v4f*)eh)[(size_t)e * 128 + c] = s;
    float* p = nm + (size_t)dst[e] * DIM + c * 4;
    atomicAdd(p + 0, fmaxf(s.x, 0.f));
    atomicAdd(p + 1, fmaxf(s.y, 0.f));
    atomicAdd(p + 2, fmaxf(s.z, 0.f));
    atomicAdd(p + 3, fmaxf(s.w, 0.f));
}

// A[e] = bf16( node_msgs[src[e]] - relu(eh[rev[e]]) )
__global__ __launch_bounds__(256) void k_build_edge_msgs(const float* __restrict__ nm,
                                                         const float* __restrict__ eh,
                                                         const int* __restrict__ src,
                                                         const int* __restrict__ rev,
                                                         __bf16* __restrict__ A) {
    int t = blockIdx.x * blockDim.x + threadIdx.x;
    int e = t >> 7, c = t & 127;
    v4f a = ((const v4f*)nm)[(size_t)src[e] * 128 + c];
    v4f b = ((const v4f*)eh)[(size_t)rev[e] * 128 + c];
    v4bf o;
    o.x = (__bf16)(a.x - fmaxf(b.x, 0.f));
    o.y = (__bf16)(a.y - fmaxf(b.y, 0.f));
    o.z = (__bf16)(a.z - fmaxf(b.z, 0.f));
    o.w = (__bf16)(a.w - fmaxf(b.w, 0.f));
    ((v4bf*)A)[(size_t)e * 128 + c] = o;
}

// Wb (bf16, [DIM,DIM] row-major, W[f][d]) from fp32 Ws layer slice
__global__ __launch_bounds__(256) void k_cvt_w(const float* __restrict__ W,
                                               __bf16* __restrict__ Wb) {
    int t = blockIdx.x * blockDim.x + threadIdx.x;   // DIM*DIM/4 threads
    v4f w = ((const v4f*)W)[t];
    v4bf o;
    o.x = (__bf16)w.x; o.y = (__bf16)w.y; o.z = (__bf16)w.z; o.w = (__bf16)w.w;
    ((v4bf*)Wb)[t] = o;
}

// ---------------- WMMA GEMM: bias + residual + fused scatter-sum ----------------
// eh[e,f] += sum_d A[e,d] * W[f,d] + bias[f]
// then: acc_out[dst[e], f] += (RELU_SCATTER ? relu(eh_new) : eh_new)
// Grid: x = column tiles (8), y = row groups (1250) -> blocks sharing A rows are
// dispatch-adjacent so A re-reads hit L2. Block = 8 waves; wave w owns rows
// [by*128 + w*16, +16) x cols [bx*64, +64).
template <bool RELU_SCATTER>
__global__ __launch_bounds__(256) void k_gemm_residual_scatter(
        const __bf16* __restrict__ A, const __bf16* __restrict__ Wb,
        const float* __restrict__ bias, const int* __restrict__ dst,
        float* __restrict__ eh, float* __restrict__ acc_out) {
    const int wave = threadIdx.x >> 5;
    const int lane = threadIdx.x & 31;
    const int half = lane >> 4;     // which 16-lane half of the wave
    const int mn   = lane & 15;     // row (for A) / col (for B & C) within tile
    const int rowBase = blockIdx.y * 128 + wave * 16;
    const int colBase = blockIdx.x * 64;

    v8f acc0 = {}, acc1 = {}, acc2 = {}, acc3 = {};

    // A fragment rows: lane holds row m = mn; K range selected by `half`
    const __bf16* arow = A + (size_t)(rowBase + mn) * DIM;
    // B fragment: B[k][n] = W[colBase + t*16 + mn][k]; contiguous along k
    const __bf16* wrow = Wb + (size_t)(colBase + mn) * DIM;

    for (int kk = 0; kk < DIM; kk += 32) {
        // A: 16x32 bf16 — lanes 0-15: K = kk..kk+15, lanes 16-31: K = kk+16..kk+31
        v16bf a = *(const v16bf*)(arow + kk + half * 16);

        // B: 32x16 bf16 — lanes 0-15: K = kk..kk+15, lanes 16-31: K = kk+16..kk+31
        const int kb = kk + half * 16;
        v16bf b0 = *(const v16bf*)(wrow + 0 * 16 * DIM + kb);
        v16bf b1 = *(const v16bf*)(wrow + 1 * 16 * DIM + kb);
        v16bf b2 = *(const v16bf*)(wrow + 2 * 16 * DIM + kb);
        v16bf b3 = *(const v16bf*)(wrow + 3 * 16 * DIM + kb);

        acc0 = __builtin_amdgcn_wmma_f32_16x16x32_bf16(false, a, false, b0, (short)0, acc0, false, false);
        acc1 = __builtin_amdgcn_wmma_f32_16x16x32_bf16(false, a, false, b1, (short)0, acc1, false, false);
        acc2 = __builtin_amdgcn_wmma_f32_16x16x32_bf16(false, a, false, b2, (short)0, acc2, false, false);
        acc3 = __builtin_amdgcn_wmma_f32_16x16x32_bf16(false, a, false, b3, (short)0, acc3, false, false);
    }

    // Epilogue: C VGPR r -> row = half*8 + r, col = mn.
    // Fuse bias + residual (fp32), then scatter-add into next-layer accumulator.
    const float b0 = bias[colBase + mn];
    const float b1 = bias[colBase + 16 + mn];
    const float b2 = bias[colBase + 32 + mn];
    const float b3 = bias[colBase + 48 + mn];
#pragma unroll
    for (int r = 0; r < 8; ++r) {
        int row = rowBase + half * 8 + r;
        float* p = eh + (size_t)row * DIM + colBase + mn;
        float v0 = p[0]  + acc0[r] + b0;
        float v1 = p[16] + acc1[r] + b1;
        float v2 = p[32] + acc2[r] + b2;
        float v3 = p[48] + acc3[r] + b3;
        p[0] = v0; p[16] = v1; p[32] = v2; p[48] = v3;
        float* q = acc_out + (size_t)dst[row] * DIM + colBase + mn;
        if (RELU_SCATTER) {
            atomicAdd(q + 0,  fmaxf(v0, 0.f));
            atomicAdd(q + 16, fmaxf(v1, 0.f));
            atomicAdd(q + 32, fmaxf(v2, 0.f));
            atomicAdd(q + 48, fmaxf(v3, 0.f));
        } else {
            atomicAdd(q + 0,  v0);
            atomicAdd(q + 16, v1);
            atomicAdd(q + 32, v2);
            atomicAdd(q + 48, v3);
        }
    }
}

// ---------------- launcher ----------------

extern "C" void kernel_launch(void* const* d_in, const int* in_sizes, int n_in,
                              void* d_out, int out_size, void* d_ws, size_t ws_size,
                              hipStream_t stream) {
    (void)in_sizes; (void)n_in; (void)out_size; (void)ws_size;

    const float* node_feats = (const float*)d_in[0];
    const float* edge_feats = (const float*)d_in[1];
    const float* Ws         = (const float*)d_in[2];   // [DEPTH, DIM, DIM]
    const float* bs         = (const float*)d_in[3];   // [DEPTH, DIM]
    const int*   edge_index = (const int*)  d_in[4];   // [2, NE]
    const int*   rev_index  = (const int*)  d_in[5];   // [NE]
    const int* src = edge_index;
    const int* dst = edge_index + NE;

    float* out_nodes = (float*)d_out;                      // [NV, DIM]
    float* eh        = (float*)d_out + (size_t)NV * DIM;   // [NE, DIM] — second output, computed in place

    // workspace: node_msgs (fp32, L2-resident) | A bf16 [NE,DIM] | Wb bf16 [DIM,DIM]
    char* ws = (char*)d_ws;
    float*  node_msgs = (float*)ws;
    __bf16* Amat      = (__bf16*)(ws + (size_t)NV * DIM * sizeof(float));
    __bf16* Wb        = (__bf16*)((char*)Amat + (size_t)NE * DIM * sizeof(__bf16));

    const dim3 blk(256);
    const int gED = (NE * 128) / 256;              // edge*float4 grid = 80000 blocks
    const int gVD = (NV * DIM / 4) / 256;          // 5000 blocks
    const dim3 gemmGrid(DIM / 64, NE / 128);       // 8 x 1250 (cols fastest -> A L2 reuse)

    // layer-0 accumulator
    k_zero<<<gVD, blk, 0, stream>>>(node_msgs, NV * DIM / 4);
    k_init_eh_scatter<<<gED, blk, 0, stream>>>(node_feats, edge_feats, src, dst, eh, node_msgs);

    for (int i = 0; i < DEPTH; ++i) {
        k_cvt_w<<<(DIM * DIM / 4) / 256, blk, 0, stream>>>(Ws + (size_t)i * DIM * DIM, Wb);
        // edge_msgs consume node_msgs of layer i ...
        k_build_edge_msgs<<<gED, blk, 0, stream>>>(node_msgs, eh, src, rev_index, Amat);
        if (i < DEPTH - 1) {
            // ... then the same buffer is recycled as the layer i+1 accumulator,
            // and the GEMM scatters relu(eh_new) into it from its epilogue.
            k_zero<<<gVD, blk, 0, stream>>>(node_msgs, NV * DIM / 4);
            k_gemm_residual_scatter<true><<<gemmGrid, blk, 0, stream>>>(
                Amat, Wb, bs + (size_t)i * DIM, dst, eh, node_msgs);
        } else {
            // last layer: scatter raw eh_new into the node-hiddens output
            k_zero<<<gVD, blk, 0, stream>>>(out_nodes, NV * DIM / 4);
            k_gemm_residual_scatter<false><<<gemmGrid, blk, 0, stream>>>(
                Amat, Wb, bs + (size_t)i * DIM, dst, eh, out_nodes);
        }
    }
}
